// MultiheadAttention_59992103191051
// MI455X (gfx1250) — compile-verified, matrix operand order checked by reading, not run on hardware
//
#include <hip/hip_runtime.h>
#include <stdint.h>

typedef __attribute__((ext_vector_type(16))) _Float16 v16h;
typedef __attribute__((ext_vector_type(8)))  float    v8f;
typedef __attribute__((ext_vector_type(4)))  int      v4i;

typedef __attribute__((address_space(1))) v4i as1_v4i;
typedef __attribute__((address_space(3))) v4i as3_v4i;

// Problem constants
static constexpr int Bc  = 4;
static constexpr int Tc  = 1024;
static constexpr int Dc  = 512;
static constexpr int Hc  = 8;
static constexpr int Kn  = 30;     // neighbors
static constexpr int EDc = 128;
static constexpr int HDc = 64;
static constexpr int Mrows = Bc * Tc;   // 4096
static constexpr int CATW = Dc + EDc;   // 640

// ---------------------------------------------------------------------------
// CDNA5 async global->LDS (ASYNCcnt-tracked). Guarded so compile never breaks.
// Builtin signature (from hipcc diagnostic): (v4i AS1*, v4i AS3*, Imm, Imm)
// ---------------------------------------------------------------------------
#if defined(__has_builtin)
#if __has_builtin(__builtin_amdgcn_global_load_async_to_lds_b128)
#define HAVE_ASYNC_LDS 1
#endif
#endif

__device__ __forceinline__ void cp16_g2l(void* lds, const void* glob) {
#ifdef HAVE_ASYNC_LDS
  __builtin_amdgcn_global_load_async_to_lds_b128(
      (as1_v4i*)(uintptr_t)glob,
      (as3_v4i*)(uint32_t)(uintptr_t)lds, 0, 0);
#else
  *(int4*)lds = *(const int4*)glob;
#endif
}

__device__ __forceinline__ void async_wait0() {
#ifdef HAVE_ASYNC_LDS
#if __has_builtin(__builtin_amdgcn_s_wait_asynccnt)
  __builtin_amdgcn_s_wait_asynccnt(0);
#else
  asm volatile("s_wait_asynccnt 0" ::: "memory");
#endif
#endif
}

// ---------------------------------------------------------------------------
// k-index map for 16-bit WMMA A/B fragments (wave32):
//   element e of v16h in lane L -> K = (e>>3)*16 + (L>>4)*8 + (e&7)
// ---------------------------------------------------------------------------
__device__ __forceinline__ int kmap(int e, int half) {
  return ((e >> 3) << 4) + half * 8 + (e & 7);
}

__device__ __forceinline__ v8f wmma_f16(v16h a, v16h b, v8f c) {
  return __builtin_amdgcn_wmma_f32_16x16x32_f16(false, a, false, b,
                                                (short)0, c, false, false);
}

// ---------------------------------------------------------------------------
// Reductions (wave32)
// ---------------------------------------------------------------------------
__device__ __forceinline__ float warp_sum(float v) {
  #pragma unroll
  for (int o = 16; o > 0; o >>= 1) v += __shfl_xor(v, o, 32);
  return v;
}
__device__ __forceinline__ float warp_max(float v) {
  #pragma unroll
  for (int o = 16; o > 0; o >>= 1) v = fmaxf(v, __shfl_xor(v, o, 32));
  return v;
}
__device__ float block_red(float v, float* sm, bool ismax) {
  v = ismax ? warp_max(v) : warp_sum(v);
  int wv = threadIdx.x >> 5, ln = threadIdx.x & 31;
  if (ln == 0) sm[wv] = v;
  __syncthreads();
  int nw = blockDim.x >> 5;
  if (threadIdx.x == 0) {
    float r = sm[0];
    for (int i = 1; i < nw; ++i) r = ismax ? fmaxf(r, sm[i]) : (r + sm[i]);
    sm[0] = r;
  }
  __syncthreads();
  float r = sm[0];
  __syncthreads();
  return r;
}

// ---------------------------------------------------------------------------
// f32 -> f16 conversion pre-pass (4-wide)
// ---------------------------------------------------------------------------
__global__ __launch_bounds__(256)
void cvt_f32_f16_kernel(const float* __restrict__ s, _Float16* __restrict__ d, int n) {
  int i = (blockIdx.x * 256 + threadIdx.x) * 4;
  if (i + 3 < n) {
    float4 v = *(const float4*)(s + i);
    d[i + 0] = (_Float16)v.x;
    d[i + 1] = (_Float16)v.y;
    d[i + 2] = (_Float16)v.z;
    d[i + 3] = (_Float16)v.w;
  }
}

// ---------------------------------------------------------------------------
// NT GEMM on f16: C[m,n] = act((sum_k A[m,k]*W[n,k] + bias[n]) * scale)
// Tiles HBM->LDS via async B128 copies; f16 WMMA, f32 accum.
// Writes f32 (Cf) and/or f16 (Ch) output.
// ---------------------------------------------------------------------------
__global__ __launch_bounds__(128)
void gemm_nt_f16_kernel(const _Float16* __restrict__ A, const _Float16* __restrict__ W,
                        const float* __restrict__ bias, float* __restrict__ Cf,
                        _Float16* __restrict__ Ch, int N, int Kdim, float scale, int act) {
  __shared__ __align__(16) _Float16 As[64][40];   // 80B row stride (16B multiple)
  __shared__ __align__(16) _Float16 Bs[64][40];
  const int tid = threadIdx.x, lane = tid & 31, wv = tid >> 5;
  const int bm = blockIdx.x * 64, bn = blockIdx.y * 64;
  const int half = lane >> 4, mr = lane & 15;
  v8f acc[4] = {};

  for (int k0 = 0; k0 < Kdim; k0 += 32) {
    for (int i = tid; i < 256; i += 128) {       // 64 rows x 4 chunks, A and B
      int r = i >> 2, c8 = (i & 3) << 3;
      cp16_g2l(&As[r][c8], A + (size_t)(bm + r) * Kdim + k0 + c8);
      cp16_g2l(&Bs[r][c8], W + (size_t)(bn + r) * Kdim + k0 + c8);
    }
    if (k0 + 32 < Kdim) {                        // warm L2 for next K tile
      __builtin_prefetch(A + (size_t)(bm + (tid >> 1)) * Kdim + k0 + 32 + (tid & 1) * 16, 0, 1);
      __builtin_prefetch(W + (size_t)(bn + (tid >> 1)) * Kdim + k0 + 32 + (tid & 1) * 16, 0, 1);
    }
    async_wait0();
    __syncthreads();
    v16h af;
    #pragma unroll
    for (int e = 0; e < 16; ++e) af[e] = As[wv * 16 + mr][kmap(e, half)];
    #pragma unroll
    for (int c = 0; c < 4; ++c) {
      v16h bf;
      #pragma unroll
      for (int e = 0; e < 16; ++e) bf[e] = Bs[c * 16 + mr][kmap(e, half)];
      acc[c] = wmma_f16(af, bf, acc[c]);
    }
    __syncthreads();
  }
  #pragma unroll
  for (int c = 0; c < 4; ++c) {
    int col = bn + c * 16 + mr;
    float bv = bias ? bias[col] : 0.0f;
    #pragma unroll
    for (int r = 0; r < 8; ++r) {
      int row = bm + wv * 16 + half * 8 + r;
      float v = (acc[c][r] + bv) * scale;
      if (act == 1) v = 0.5f * v * (1.0f + erff(v * 0.70710678118f));  // exact GELU
      if (Cf) Cf[(size_t)row * N + col] = v;
      if (Ch) Ch[(size_t)row * N + col] = (_Float16)v;
    }
  }
}

// ---------------------------------------------------------------------------
// logits[b,h,t,s] = clamp((q·k + bias) masked, * 4^-tanh(t0)) -> f16
// ---------------------------------------------------------------------------
__global__ __launch_bounds__(128)
void scores_kernel(const _Float16* __restrict__ q, const _Float16* __restrict__ kk,
                   const float* __restrict__ bias, const int* __restrict__ mask,
                   const float* __restrict__ tscal, _Float16* __restrict__ sc) {
  const int st = blockIdx.x, qt = blockIdx.y, bh = blockIdx.z;
  const int b = bh >> 3, h = bh & 7;
  __shared__ __align__(16) _Float16 Qs[64][40];
  __shared__ __align__(16) _Float16 Ks[64][40];
  const int tid = threadIdx.x, lane = tid & 31, wv = tid >> 5;
  const int half = lane >> 4, mr = lane & 15;
  v8f acc[4] = {};
  const size_t qbase = ((size_t)b * Tc + qt * 64) * Dc + h * HDc;
  const size_t kbase = ((size_t)b * Tc + st * 64) * Dc + h * HDc;

  for (int k0 = 0; k0 < HDc; k0 += 32) {
    for (int i = tid; i < 256; i += 128) {
      int r = i >> 2, c8 = (i & 3) << 3;
      cp16_g2l(&Qs[r][c8], q  + qbase + (size_t)r * Dc + k0 + c8);
      cp16_g2l(&Ks[r][c8], kk + kbase + (size_t)r * Dc + k0 + c8);
    }
    async_wait0();
    __syncthreads();
    v16h af;
    #pragma unroll
    for (int e = 0; e < 16; ++e) af[e] = Qs[wv * 16 + mr][kmap(e, half)];
    #pragma unroll
    for (int c = 0; c < 4; ++c) {
      v16h bf;
      #pragma unroll
      for (int e = 0; e < 16; ++e) bf[e] = Ks[c * 16 + mr][kmap(e, half)];
      acc[c] = wmma_f16(af, bf, acc[c]);
    }
    __syncthreads();
  }
  const float it = exp2f(-2.0f * tanhf(tscal[0]));  // 1 / 4^tanh(t)
  #pragma unroll
  for (int c = 0; c < 4; ++c) {
    int s = st * 64 + c * 16 + mr;
    int mk = mask[b * Tc + s];
    #pragma unroll
    for (int r = 0; r < 8; ++r) {
      int trow = qt * 64 + wv * 16 + half * 8 + r;
      float v = acc[c][r] + bias[((size_t)b * Tc + trow) * Tc + s];
      if (mk == 0) v = -1e9f;
      v = fmaxf(v * it, -60000.0f);               // keep representable in f16
      sc[((size_t)bh * Tc + trow) * Tc + s] = (_Float16)v;
    }
  }
}

// ---------------------------------------------------------------------------
// Row softmax (f32 math on f16 logits, in-place f16 p) + head-mean ma (f32)
// ---------------------------------------------------------------------------
__global__ __launch_bounds__(256)
void softmax_ma_kernel(_Float16* __restrict__ sc, float* __restrict__ ma) {
  __shared__ float sm[8];
  const int bt = blockIdx.x;
  const int b = bt >> 10, t = bt & 1023;
  const int tid = threadIdx.x;
  float maacc[4] = {0.f, 0.f, 0.f, 0.f};
  for (int h = 0; h < Hc; ++h) {
    _Float16* row = sc + ((size_t)(b * Hc + h) * Tc + t) * Tc;
    float vals[4];
    float mx = -3.4e38f;
    #pragma unroll
    for (int j = 0; j < 4; ++j) { vals[j] = (float)row[tid + j * 256]; mx = fmaxf(mx, vals[j]); }
    mx = block_red(mx, sm, true);
    float s = 0.f;
    #pragma unroll
    for (int j = 0; j < 4; ++j) { vals[j] = expf(vals[j] - mx); s += vals[j]; }
    s = block_red(s, sm, false);
    float inv = 1.0f / s;
    #pragma unroll
    for (int j = 0; j < 4; ++j) {
      float p = vals[j] * inv;
      row[tid + j * 256] = (_Float16)p;
      maacc[j] += p * 0.125f;
    }
  }
  #pragma unroll
  for (int j = 0; j < 4; ++j)
    ma[(size_t)bt * Tc + tid + j * 256] = maacc[j];
}

// ---------------------------------------------------------------------------
// attn = p·v (NN GEMM, K=T), into first D columns of f16 concat buffer
// ---------------------------------------------------------------------------
__global__ __launch_bounds__(128)
void pv_kernel(const _Float16* __restrict__ p, const _Float16* __restrict__ vv,
               _Float16* __restrict__ cat) {
  const int qt = blockIdx.x, bh = blockIdx.y;
  const int b = bh >> 3, h = bh & 7;
  __shared__ __align__(16) _Float16 Ps[64][40];
  __shared__ __align__(16) _Float16 Vs[32][72];   // 144B row stride (16B multiple)
  const int tid = threadIdx.x, lane = tid & 31, wv = tid >> 5;
  const int half = lane >> 4, mr = lane & 15;
  v8f acc[4] = {};
  const _Float16* prow = p + ((size_t)bh * Tc + qt * 64) * Tc;
  const _Float16* vb   = vv + (size_t)b * Tc * Dc + h * HDc;

  for (int k0 = 0; k0 < Tc; k0 += 32) {
    for (int i = tid; i < 256; i += 128) {       // P: 64 rows x 4 chunks
      int r = i >> 2, c8 = (i & 3) << 3;
      cp16_g2l(&Ps[r][c8], prow + (size_t)r * Tc + k0 + c8);
    }
    for (int i = tid; i < 256; i += 128) {       // V: 32 rows x 8 chunks
      int r = i >> 3, c8 = (i & 7) << 3;
      cp16_g2l(&Vs[r][c8], vb + (size_t)(k0 + r) * Dc + c8);
    }
    async_wait0();
    __syncthreads();
    v16h af;
    #pragma unroll
    for (int e = 0; e < 16; ++e) af[e] = Ps[wv * 16 + mr][kmap(e, half)];
    #pragma unroll
    for (int c = 0; c < 4; ++c) {
      v16h bf;
      #pragma unroll
      for (int e = 0; e < 16; ++e) bf[e] = Vs[kmap(e, half)][c * 16 + mr];
      acc[c] = wmma_f16(af, bf, acc[c]);
    }
    __syncthreads();
  }
  #pragma unroll
  for (int c = 0; c < 4; ++c) {
    int col = h * HDc + c * 16 + mr;
    #pragma unroll
    for (int r = 0; r < 8; ++r) {
      int trow = qt * 64 + wv * 16 + half * 8 + r;
      cat[((size_t)b * Tc + trow) * CATW + col] = (_Float16)acc[c][r];
    }
  }
}

// ---------------------------------------------------------------------------
// edge[b,t,:] = sum_k g_k*E[b,t,k,:]/(sum g + 1e-6), g = ma at E_idx
// ---------------------------------------------------------------------------
__global__ __launch_bounds__(128)
void edge_kernel(const float* __restrict__ ma, const int* __restrict__ E_idx,
                 const float* __restrict__ E, _Float16* __restrict__ cat) {
  __shared__ float gsh[32];
  __shared__ float gsum;
  const int bt = blockIdx.x;
  const int tid = threadIdx.x;
  if (tid < Kn) gsh[tid] = ma[(size_t)bt * Tc + E_idx[(size_t)bt * Kn + tid]];
  __syncthreads();
  if (tid == 0) {
    float s = 0.f;
    for (int k = 0; k < Kn; ++k) s += gsh[k];
    gsum = s + 1e-6f;
  }
  __syncthreads();
  float acc = 0.f;
  for (int k = 0; k < Kn; ++k)
    acc += gsh[k] * E[((size_t)bt * Kn + k) * EDc + tid];
  cat[(size_t)bt * CATW + Dc + tid] = (_Float16)(acc / gsum);
}

// ---------------------------------------------------------------------------
// out = LayerNorm(a + b), width 512; optional f16 copy of the output
// ---------------------------------------------------------------------------
__global__ __launch_bounds__(256)
void add_ln_kernel(const float* __restrict__ a, const float* __restrict__ bsrc,
                   const float* __restrict__ g, const float* __restrict__ be,
                   float eps, float* __restrict__ out, _Float16* __restrict__ out16) {
  __shared__ float sm[8];
  const size_t base = (size_t)blockIdx.x * Dc;
  const int tid = threadIdx.x;
  float x0 = a[base + tid] + bsrc[base + tid];
  float x1 = a[base + 256 + tid] + bsrc[base + 256 + tid];
  float s = block_red(x0 + x1, sm, false);
  float m = s * (1.0f / (float)Dc);
  float d0 = x0 - m, d1 = x1 - m;
  float sq = block_red(d0 * d0 + d1 * d1, sm, false);
  float inv = rsqrtf(sq * (1.0f / (float)Dc) + eps);
  float y0 = d0 * inv * g[tid] + be[tid];
  float y1 = d1 * inv * g[256 + tid] + be[256 + tid];
  out[base + tid] = y0;
  out[base + 256 + tid] = y1;
  if (out16) {
    out16[base + tid] = (_Float16)y0;
    out16[base + 256 + tid] = (_Float16)y1;
  }
}

// ---------------------------------------------------------------------------
extern "C" void kernel_launch(void* const* d_in, const int* in_sizes, int n_in,
                              void* d_out, int out_size, void* d_ws, size_t ws_size,
                              hipStream_t stream) {
  const float* h_V  = (const float*)d_in[0];
  const float* kin  = (const float*)d_in[1];
  const float* vin  = (const float*)d_in[2];
  const float* E    = (const float*)d_in[3];
  const float* bias = (const float*)d_in[4];
  const float* Wq   = (const float*)d_in[5];
  const float* bq   = (const float*)d_in[6];
  const float* Wk   = (const float*)d_in[7];
  const float* bk   = (const float*)d_in[8];
  const float* Wv   = (const float*)d_in[9];
  const float* bv   = (const float*)d_in[10];
  const float* Wo   = (const float*)d_in[11];
  const float* bo   = (const float*)d_in[12];
  const float* tsc  = (const float*)d_in[13];
  const float* gn   = (const float*)d_in[14];
  const float* bn   = (const float*)d_in[15];
  const float* w1   = (const float*)d_in[16];
  const float* bw1  = (const float*)d_in[17];
  const float* w2   = (const float*)d_in[18];
  const float* bw2  = (const float*)d_in[19];
  const float* gf   = (const float*)d_in[20];
  const float* bf   = (const float*)d_in[21];
  const float* g2   = (const float*)d_in[22];
  const float* b2   = (const float*)d_in[23];
  const int*  E_idx = (const int*)d_in[24];
  const int*  mask  = (const int*)d_in[25];
  float* outp = (float*)d_out;

  // ---- workspace carve-up (256B-aligned) ----
  size_t off = 0;
  auto alloc = [&](size_t bytes) {
    void* p = (char*)d_ws + off;
    off += (bytes + 255) & ~(size_t)255;
    return p;
  };
  const size_t MD = (size_t)Mrows * Dc;                 // 2,097,152 elems
  _Float16* x16   = (_Float16*)alloc(MD * 2);
  _Float16* kin16 = (_Float16*)alloc(MD * 2);
  _Float16* vin16 = (_Float16*)alloc(MD * 2);
  _Float16* Wq16  = (_Float16*)alloc((size_t)Dc * Dc * 2);
  _Float16* Wk16  = (_Float16*)alloc((size_t)Dc * Dc * 2);
  _Float16* Wv16  = (_Float16*)alloc((size_t)Dc * Dc * 2);
  _Float16* Wo16  = (_Float16*)alloc((size_t)Dc * CATW * 2);
  _Float16* w116  = (_Float16*)alloc((size_t)4 * Dc * Dc * 2);
  _Float16* w216  = (_Float16*)alloc((size_t)4 * Dc * Dc * 2);
  _Float16* q16   = (_Float16*)alloc(MD * 2);
  _Float16* kp16  = (_Float16*)alloc(MD * 2);
  _Float16* vp16  = (_Float16*)alloc(MD * 2);
  _Float16* cat16 = (_Float16*)alloc((size_t)Mrows * CATW * 2);
  float*    ma    = (float*)   alloc((size_t)Bc * Tc * Tc * 4);
  float*    proj  = (float*)   alloc(MD * 4);
  float*    hf    = (float*)   alloc(MD * 4);
  _Float16* h16   = (_Float16*)alloc(MD * 2);
  _Float16* sc16  = (_Float16*)alloc((size_t)Bc * Hc * Tc * Tc * 2);  // 64 MiB; logits then p
  // FFN temporaries alias the (dead after pv) scores/p region
  _Float16* ff116 = sc16;                                       // 16 MiB
  float*    ff2   = (float*)((char*)sc16 + ((size_t)32 << 20)); // +32 MiB, 8 MiB
  float*    dh    = (float*)((char*)sc16 + ((size_t)48 << 20)); // +48 MiB, 8 MiB

  const float SCALE = 0.125f;  // HD^-0.5
  dim3 blk128(128), blk256(256);

  // 0: f32 -> f16 conversions (inputs + all GEMM weights)
  auto cvt = [&](const float* s, _Float16* d, size_t n) {
    cvt_f32_f16_kernel<<<dim3((unsigned)(n / 1024)), blk256, 0, stream>>>(s, d, (int)n);
  };
  cvt(h_V, x16, MD);
  cvt(kin, kin16, MD);
  cvt(vin, vin16, MD);
  cvt(Wq, Wq16, (size_t)Dc * Dc);
  cvt(Wk, Wk16, (size_t)Dc * Dc);
  cvt(Wv, Wv16, (size_t)Dc * Dc);
  cvt(Wo, Wo16, (size_t)Dc * CATW);
  cvt(w1, w116, (size_t)4 * Dc * Dc);
  cvt(w2, w216, (size_t)4 * Dc * Dc);

  // 1-3: QKV projections (q folds the softmax scale), f16 outputs
  gemm_nt_f16_kernel<<<dim3(Mrows / 64, Dc / 64), blk128, 0, stream>>>(x16, Wq16, bq, nullptr, q16, Dc, Dc, SCALE, 0);
  gemm_nt_f16_kernel<<<dim3(Mrows / 64, Dc / 64), blk128, 0, stream>>>(kin16, Wk16, bk, nullptr, kp16, Dc, Dc, 1.0f, 0);
  gemm_nt_f16_kernel<<<dim3(Mrows / 64, Dc / 64), blk128, 0, stream>>>(vin16, Wv16, bv, nullptr, vp16, Dc, Dc, 1.0f, 0);

  // 4: attention logits (f16)
  scores_kernel<<<dim3(Tc / 64, Tc / 64, Bc * Hc), blk128, 0, stream>>>(q16, kp16, bias, mask, tsc, sc16);

  // 5: softmax (in-place f16 p) + head-mean ma
  softmax_ma_kernel<<<dim3(Bc * Tc), blk256, 0, stream>>>(sc16, ma);

  // 6: p·v -> attn half of concat (f16)
  pv_kernel<<<dim3(Tc / 64, Bc * Hc), blk128, 0, stream>>>(sc16, vp16, cat16);

  // 7: edge gather -> edge half of concat (f16)
  edge_kernel<<<dim3(Bc * Tc), blk128, 0, stream>>>(ma, E_idx, E, cat16);

  // 8: output projection (K = 640) -> f32
  gemm_nt_f16_kernel<<<dim3(Mrows / 64, Dc / 64), blk128, 0, stream>>>(cat16, Wo16, bo, proj, nullptr, Dc, CATW, 1.0f, 0);

  // 9: h = LN(h_V + proj), eps 1e-5 (f32 + f16 copy)
  add_ln_kernel<<<dim3(Mrows), blk256, 0, stream>>>(h_V, proj, gn, bn, 1e-5f, hf, h16);

  // 10: FFN1 + exact GELU (N=2048) -> f16
  gemm_nt_f16_kernel<<<dim3(Mrows / 64, (4 * Dc) / 64), blk128, 0, stream>>>(h16, w116, bw1, nullptr, ff116, 4 * Dc, Dc, 1.0f, 1);

  // 11: FFN2 (K=2048) -> f32
  gemm_nt_f16_kernel<<<dim3(Mrows / 64, Dc / 64), blk128, 0, stream>>>(ff116, w216, bw2, ff2, nullptr, Dc, 4 * Dc, 1.0f, 0);

  // 12: dh = LN(h + ff, eps 1e-6)
  add_ln_kernel<<<dim3(Mrows), blk256, 0, stream>>>(hf, ff2, gf, bf, 1e-6f, dh, nullptr);

  // 13: out = LN(h + dh, eps 1e-5)
  add_ln_kernel<<<dim3(Mrows), blk256, 0, stream>>>(hf, dh, g2, b2, 1e-5f, outp, nullptr);
}
